// GRU_67542655696942
// MI455X (gfx1250) — matrix operand order repaired and checked by reading, not compiled, not run on hardware
//
#include <hip/hip_runtime.h>

// ---------------------------------------------------------------------------
// GRU on MI455X (gfx1250).
//  Kernel 1: fused f16-WMMA GEMM for xz/xr/xh = x @ W_x + bias  (206 GFLOP)
//            Staging uses packed v_cvt_pk_rtz_f16_f32 + ds_store_b32.
//  Kernel 2: persistent recurrent kernel, 32 blocks, weights resident in LDS
//            (3 x 32cols x 1024K fp16, ~193KB of the 320KB WGP LDS),
//            two WMMA phases + two grid syncs per time step, h kept in fp32.
// ---------------------------------------------------------------------------

#define BSZ   64
#define TT    512
#define FEAT  1024
#define HID   1024
#define NBLK  32          // hidden columns per persistent block
#define KP    1032        // padded K pitch (halves): 2064B = 129*16B, bank-conflict free

typedef __fp16   v2hf __attribute__((ext_vector_type(2)));   // type returned by cvt_pkrtz
typedef _Float16 v8h  __attribute__((ext_vector_type(8)));
typedef _Float16 v16h __attribute__((ext_vector_type(16)));
typedef float    v8f  __attribute__((ext_vector_type(8)));

#define SHUF16(lo, hi) __builtin_shufflevector((lo), (hi), 0,1,2,3,4,5,6,7,8,9,10,11,12,13,14,15)
#define WMMA(a, b, c)  __builtin_amdgcn_wmma_f32_16x16x32_f16(false, (a), false, (b), (short)0, (c), false, false)

__device__ __forceinline__ v8f zero8() {
    v8f z = {0.0f,0.0f,0.0f,0.0f,0.0f,0.0f,0.0f,0.0f};
    return z;
}

// one v_cvt_pk_rtz_f16_f32 + one ds_store_b32 (dst must be 4B aligned)
__device__ __forceinline__ void store_pk(_Float16* dst, float a, float b) {
    *(v2hf*)dst = __builtin_amdgcn_cvt_pkrtz(a, b);   // bit-identical fp16 pair
}

__device__ __forceinline__ void grid_sync(unsigned* cnt, unsigned& gen, unsigned nblk) {
    __syncthreads();
    if (threadIdx.x == 0) {
        gen += nblk;
        __hip_atomic_fetch_add(cnt, 1u, __ATOMIC_RELEASE, __HIP_MEMORY_SCOPE_AGENT);
        while (__hip_atomic_load(cnt, __ATOMIC_ACQUIRE, __HIP_MEMORY_SCOPE_AGENT) < gen) {
            __builtin_amdgcn_s_sleep(2);
        }
    }
    __syncthreads();
}

// ---------------------------------------------------------------------------
// Kernel 1: XZ/XR/XH[t][b][n] (fp16) = x[b][t][:] @ Wg_x + bias_g
// grid (16 nTiles, 512 mTiles, 3 gates), 256 threads (8 waves).
// Block tile 64x64, K-step 32, A(x) row-major LDS, B(W) transposed LDS.
// ---------------------------------------------------------------------------
__global__ __launch_bounds__(256) void gru_pre_gemm(
    const float* __restrict__ x,
    const float* __restrict__ wz, const float* __restrict__ wr, const float* __restrict__ ww,
    const float* __restrict__ bz, const float* __restrict__ br, const float* __restrict__ bh,
    _Float16* __restrict__ xz, _Float16* __restrict__ xr, _Float16* __restrict__ xh)
{
    __shared__ _Float16 XT[64 * 40];   // [m][k], pitch 40 halves (80B, 16B aligned)
    __shared__ _Float16 WT[64 * 40];   // [n][k]

    const int g = blockIdx.z;
    const float* wsrc = (g == 0 ? wz : (g == 1 ? wr : ww)) + (size_t)HID * HID; // x-part rows [H, H+F)
    const float* bias = (g == 0 ? bz : (g == 1 ? br : bh));
    _Float16* outp    = (g == 0 ? xz : (g == 1 ? xr : xh));

    const int n0  = blockIdx.x * 64;
    const int m0  = blockIdx.y * 64;        // global M = b*T + t
    const int b   = m0 >> 9;                // T = 512
    const int tb  = m0 & 511;
    const int tid = threadIdx.x, lane = tid & 31, wave = tid >> 5;
    const int mt  = wave & 3;               // M-tile of this wave
    const int ntb = (wave >> 2) * 2;        // two N-tiles per wave
    const int l16 = lane & 15, lh = lane >> 4;

    v8f acc0 = zero8(), acc1 = zero8();

    for (int kb = 0; kb < FEAT; kb += 32) {
        #pragma unroll
        for (int r = 0; r < 4; ++r) {       // x tile 64x32: float2 -> pk-f16 -> b32
            int idx = r * 256 + tid;
            int i = idx >> 4, kp = (idx & 15) * 2;
            const float* src = &x[(size_t)(m0 + i) * FEAT + kb + kp];
            store_pk(&XT[i * 40 + kp], src[0], src[1]);
        }
        #pragma unroll
        for (int r = 0; r < 4; ++r) {       // W tile 32x64 -> transposed [n][k], packed pairs
            int idx = r * 256 + tid;
            int n = idx & 63, kp = ((idx >> 6) & 15) * 2;
            const float* src = &wsrc[(size_t)(kb + kp) * HID + n0 + n];
            store_pk(&WT[n * 40 + kp], src[0], src[HID]);
        }
        __syncthreads();

        const _Float16* ap = &XT[(mt * 16 + l16) * 40 + lh * 8];
        v16h a = SHUF16(*(const v8h*)ap, *(const v8h*)(ap + 16));

        const _Float16* bp0 = &WT[((ntb + 0) * 16 + l16) * 40 + lh * 8];
        v16h b0 = SHUF16(*(const v8h*)bp0, *(const v8h*)(bp0 + 16));
        acc0 = WMMA(a, b0, acc0);

        const _Float16* bp1 = &WT[((ntb + 1) * 16 + l16) * 40 + lh * 8];
        v16h b1 = SHUF16(*(const v8h*)bp1, *(const v8h*)(bp1 + 16));
        acc1 = WMMA(a, b1, acc1);

        __syncthreads();
    }

    #pragma unroll
    for (int j = 0; j < 2; ++j) {
        v8f acc = j ? acc1 : acc0;
        int ng = n0 + (ntb + j) * 16 + l16;
        float bv = bias[ng];
        #pragma unroll
        for (int gg = 0; gg < 8; ++gg) {
            int m = mt * 16 + gg + lh * 8;  // C layout: row = g + 8*(lane>=16)
            int t = tb + m;
            outp[((size_t)t * BSZ + b) * HID + ng] = (_Float16)(acc[gg] + bv);
        }
    }
}

// ---------------------------------------------------------------------------
// Kernel 2: persistent recurrent scan. 32 blocks x 256 threads.
// Phase 1: Z = sigmoid(h@Wz + xz), R = sigmoid(h@Wr + xr); publish rh = R*h (f16).
// Phase 2: Hbar = tanh(rh@Wh + xh); h = (1-Z)*h + Z*Hbar; publish h (f16 + f32 out).
// ---------------------------------------------------------------------------
__global__ __launch_bounds__(256) void gru_recurrent(
    const float* __restrict__ h0,
    const float* __restrict__ wz, const float* __restrict__ wr, const float* __restrict__ ww,
    const _Float16* __restrict__ xz, const _Float16* __restrict__ xr, const _Float16* __restrict__ xh,
    _Float16* __restrict__ h16, _Float16* __restrict__ rh16,
    unsigned* __restrict__ bar, float* __restrict__ out)
{
    extern __shared__ _Float16 smem[];
    _Float16* WzT = smem;                    // [NBLK][KP]
    _Float16* WrT = WzT + NBLK * KP;
    _Float16* WhT = WrT + NBLK * KP;
    float* z_lds  = (float*)(WhT + NBLK * KP);  // [64][33]
    float* h_lds  = z_lds + 64 * 33;            // [64][33] fp32 master copy of h slice

    const int n0  = blockIdx.x * NBLK;
    const int tid = threadIdx.x, lane = tid & 31, wave = tid >> 5;
    const int l16 = lane & 15, lh = lane >> 4;
    const int mt  = wave & 3;
    const int gatew = wave >> 2;             // phase1: 0 -> Z-gate, 1 -> R-gate

    // ---- startup: stage recurrent weights (h-part rows [0,H)) into LDS, transposed.
    // Packed K-pairs: 2 strided f32 loads -> v_cvt_pk_rtz -> one ds_store_b32.
    for (int g = 0; g < 3; ++g) {
        const float* src = (g == 0 ? wz : (g == 1 ? wr : ww));
        _Float16* dst    = (g == 0 ? WzT : (g == 1 ? WrT : WhT));
        for (int r = 0; r < 64; ++r) {
            int idx = r * 256 + tid;
            int n = idx & 31, kp = (idx >> 5) * 2;
            const float* s = &src[(size_t)kp * HID + n0 + n];
            store_pk(&dst[n * KP + kp], s[0], s[HID]);
        }
    }
    for (int r = 0; r < 8; ++r) {            // init h (fp32 in LDS, fp16 slice to global)
        int idx = r * 256 + tid;
        int bb = idx >> 5, n = idx & 31;
        float hv = h0[(size_t)bb * HID + n0 + n];
        h_lds[bb * 33 + n] = hv;
        h16[(size_t)bb * HID + n0 + n] = (_Float16)hv;
    }
    unsigned gen = 0;
    grid_sync(bar, gen, gridDim.x);

    for (int t = 0; t < TT; ++t) {
        // ---------------- phase 1: Z and R gates ----------------
        {
            const _Float16* WT = gatew ? WrT : WzT;
            v8f acc0 = zero8(), acc1 = zero8();
            const _Float16* agp = h16 + (size_t)(mt * 16 + l16) * HID + lh * 8;
            #pragma unroll 4
            for (int ks = 0; ks < 32; ++ks) {
                const int kb = ks * 32;
                v16h a = SHUF16(*(const v8h*)(agp + kb), *(const v8h*)(agp + kb + 16));
                const _Float16* bp0 = &WT[(0 * 16 + l16) * KP + kb + lh * 8];
                v16h b0 = SHUF16(*(const v8h*)bp0, *(const v8h*)(bp0 + 16));
                const _Float16* bp1 = &WT[(1 * 16 + l16) * KP + kb + lh * 8];
                v16h b1 = SHUF16(*(const v8h*)bp1, *(const v8h*)(bp1 + 16));
                acc0 = WMMA(a, b0, acc0);
                acc1 = WMMA(a, b1, acc1);
            }
            const _Float16* xg = gatew ? xr : xz;
            #pragma unroll
            for (int j = 0; j < 2; ++j) {
                v8f acc = j ? acc1 : acc0;
                int nl = j * 16 + l16, ng = n0 + nl;
                #pragma unroll
                for (int gg = 0; gg < 8; ++gg) {
                    int bb = mt * 16 + gg + lh * 8;
                    float pre = acc[gg] + (float)xg[((size_t)t * BSZ + bb) * HID + ng];
                    float s = 1.0f / (1.0f + __expf(-pre));
                    if (gatew == 0) z_lds[bb * 33 + nl] = s;
                    else            rh16[(size_t)bb * HID + ng] = (_Float16)(s * h_lds[bb * 33 + nl]);
                }
            }
        }
        grid_sync(bar, gen, gridDim.x);

        // ---------------- phase 2: Hbar + state update ----------------
        {
            const int nt = wave >> 2;
            v8f acc = zero8();
            const _Float16* agp = rh16 + (size_t)(mt * 16 + l16) * HID + lh * 8;
            #pragma unroll 4
            for (int ks = 0; ks < 32; ++ks) {
                const int kb = ks * 32;
                v16h a = SHUF16(*(const v8h*)(agp + kb), *(const v8h*)(agp + kb + 16));
                const _Float16* bp = &WhT[(nt * 16 + l16) * KP + kb + lh * 8];
                v16h bmat = SHUF16(*(const v8h*)bp, *(const v8h*)(bp + 16));
                acc = WMMA(a, bmat, acc);
            }
            int nl = nt * 16 + l16, ng = n0 + nl;
            #pragma unroll
            for (int gg = 0; gg < 8; ++gg) {
                int bb = mt * 16 + gg + lh * 8;
                float pre  = acc[gg] + (float)xh[((size_t)t * BSZ + bb) * HID + ng];
                float e    = __expf(2.0f * pre);          // tanh
                float hbar = (e - 1.0f) / (e + 1.0f);
                float z    = z_lds[bb * 33 + nl];
                float hv   = h_lds[bb * 33 + nl];
                float hn   = (1.0f - z) * hv + z * hbar;
                out[((size_t)bb * TT + t) * HID + ng] = hn;   // output (BS,T,H)
                h_lds[bb * 33 + nl] = hn;
                h16[(size_t)bb * HID + ng] = (_Float16)hn;
                if (t == TT - 1)
                    out[(size_t)BSZ * TT * HID + (size_t)bb * HID + ng] = hn;  // h_n
            }
        }
        grid_sync(bar, gen, gridDim.x);
    }
}

// ---------------------------------------------------------------------------
// Workspace layout (bytes)
// ---------------------------------------------------------------------------
static constexpr size_t XELEMS   = (size_t)TT * BSZ * HID;       // 33,554,432
static constexpr size_t OFF_XZ   = 0;
static constexpr size_t OFF_XR   = XELEMS * 2;
static constexpr size_t OFF_XH   = 2 * XELEMS * 2;
static constexpr size_t OFF_H16  = 3 * XELEMS * 2;               // ~201 MB
static constexpr size_t OFF_RH16 = OFF_H16 + (size_t)BSZ * HID * 2;
static constexpr size_t OFF_CNT  = OFF_RH16 + (size_t)BSZ * HID * 2;

extern "C" void kernel_launch(void* const* d_in, const int* in_sizes, int n_in,
                              void* d_out, int out_size, void* d_ws, size_t ws_size,
                              hipStream_t stream) {
    (void)in_sizes; (void)n_in; (void)out_size; (void)ws_size;
    const float* x  = (const float*)d_in[0];
    const float* h0 = (const float*)d_in[1];
    const float* wz = (const float*)d_in[2];
    const float* wr = (const float*)d_in[3];
    const float* ww = (const float*)d_in[4];
    const float* bz = (const float*)d_in[5];
    const float* br = (const float*)d_in[6];
    const float* bh = (const float*)d_in[7];
    float* out = (float*)d_out;

    char* ws = (char*)d_ws;
    _Float16* xz   = (_Float16*)(ws + OFF_XZ);
    _Float16* xr   = (_Float16*)(ws + OFF_XR);
    _Float16* xh   = (_Float16*)(ws + OFF_XH);
    _Float16* h16  = (_Float16*)(ws + OFF_H16);
    _Float16* rh16 = (_Float16*)(ws + OFF_RH16);
    unsigned* bar  = (unsigned*)(ws + OFF_CNT);

    // reset grid-sync counter (graph-capture-safe stream op)
    (void)hipMemsetAsync((void*)bar, 0, 256, stream);

    // Kernel 1: input projections
    dim3 g1(HID / 64, (BSZ * TT) / 64, 3);
    gru_pre_gemm<<<g1, dim3(256), 0, stream>>>(x, wz, wr, ww, bz, br, bh, xz, xr, xh);

    // Kernel 2: persistent recurrent scan (210 KB dynamic LDS per block)
    size_t ldsB = (size_t)3 * NBLK * KP * sizeof(_Float16) + (size_t)2 * 64 * 33 * sizeof(float);
    (void)hipFuncSetAttribute(reinterpret_cast<const void*>(&gru_recurrent),
                              hipFuncAttributeMaxDynamicSharedMemorySize, (int)ldsB);
    gru_recurrent<<<dim3(HID / NBLK), dim3(256), ldsB, stream>>>(
        h0, wz, wr, ww, xz, xr, xh, h16, rh16, bar, out);
}
